// MultiHeadAttentionCombinedQKV_89816356094509
// MI455X (gfx1250) — matrix operand order repaired and checked by reading, not compiled, not run on hardware
//
#include <hip/hip_runtime.h>
#include <hip/hip_bf16.h>

#define B_  2
#define S_  2048
#define D_  1024
#define H_  16
#define HD_ 64
#define M_  (B_ * S_)      // 4096 rows

typedef __attribute__((ext_vector_type(16))) __bf16 bf16x16;
typedef __attribute__((ext_vector_type(8)))  __bf16 bf16x8;
typedef __attribute__((ext_vector_type(8)))  float  f32x8;
typedef __attribute__((ext_vector_type(4)))  float  f32x4;

// log2(e)/sqrt(HD): folded into Q so softmax runs in base-2 (single v_exp_f32)
#define QSCALE 0.18033688011112042f

__device__ __forceinline__ __bf16 f2bf(float f) { return (__bf16)f; }

__device__ __forceinline__ f32x8 zero8() {
  f32x8 z;
#pragma unroll
  for (int i = 0; i < 8; ++i) z[i] = 0.0f;
  return z;
}

// ---- WMMA operand loaders (wave32, v_wmma_f32_16x16x32_bf16) ----
// A operand (16x32): lane holds row=lane&15; elems[0..7]=K(8*hi..+7),
// elems[8..15]=K(16+8*hi..+7)
__device__ __forceinline__ bf16x16 load_a_bf16(const __bf16* base, int ld, int lane) {
  int row = lane & 15, hi = lane >> 4;
  const bf16x8* p0 = (const bf16x8*)(base + (size_t)row * ld + 8 * hi);
  const bf16x8* p1 = (const bf16x8*)(base + (size_t)row * ld + 16 + 8 * hi);
  bf16x8 lo = *p0, hv = *p1;
  bf16x16 a;
#pragma unroll
  for (int i = 0; i < 8; ++i) { a[i] = lo[i]; a[8 + i] = hv[i]; }
  return a;
}

// B operand (32x16) loaded from B^T row-major: lane holds col n=lane&15,
// elems[0..7]=K(8*hi..+7), elems[8..15]=K(16+8*hi..+7).
__device__ __forceinline__ bf16x16 load_b_bf16(const __bf16* base, int ld, int lane) {
  int row = lane & 15, hi = lane >> 4;
  const bf16x8* p = (const bf16x8*)(base + (size_t)row * ld + 8 * hi);
  bf16x8 lo = p[0], hv = p[2];
  bf16x16 b;
#pragma unroll
  for (int i = 0; i < 8; ++i) { b[i] = lo[i]; b[8 + i] = hv[i]; }
  return b;
}

__device__ __forceinline__ f32x8 wmma_bf16(bf16x16 a, bf16x16 b, f32x8 c) {
  return __builtin_amdgcn_wmma_f32_16x16x32_bf16(false, a, false, b, (short)0, c,
                                                 false, false);
}

// =============================================================================
// Kernel 0: fp32 -> bf16 staging (memory bound; keeps GEMM K-loops VALU-free)
// =============================================================================
__global__ __launch_bounds__(256) void cvt_kernel(const float* __restrict__ src,
                                                  __bf16* __restrict__ dst, int n8) {
  int i = blockIdx.x * 256 + threadIdx.x;
  if (i >= n8) return;
  const f32x4* p = (const f32x4*)src + 2 * (size_t)i;
  f32x4 a = p[0], b = p[1];
  bf16x8 o;
#pragma unroll
  for (int j = 0; j < 4; ++j) { o[j] = f2bf(a[j]); o[4 + j] = f2bf(b[j]); }
  ((bf16x8*)dst)[i] = o;
}

// =============================================================================
// Kernel 1: QKV projection. One wave computes a 32(M) x 64(N) tile:
// 8 WMMAs per 32-wide K step against 6 b128-pair loads.
// Scatter to Q[b,h,s,hd] (prescaled by log2e/8), K[b,h,s,hd], Vt[b,h,hd,s].
// =============================================================================
__global__ __launch_bounds__(256) void qkv_gemm_kernel(
    const __bf16* __restrict__ xbf, const __bf16* __restrict__ wbf,
    __bf16* __restrict__ Q, __bf16* __restrict__ K, __bf16* __restrict__ Vt) {
  const int lane = threadIdx.x & 31;
  const int W = blockIdx.x * 8 + (threadIdx.x >> 5);
  const int m0 = (W & 127) * 32;   // 128 m-tiles of 32 rows
  const int n0 = (W >> 7) * 64;    // 48 n-groups of 64 cols
  f32x8 acc[2][4];
#pragma unroll
  for (int i = 0; i < 2; ++i)
#pragma unroll
    for (int t = 0; t < 4; ++t) acc[i][t] = zero8();

  const __bf16* a0 = xbf + (size_t)m0 * D_;
  const __bf16* a1 = a0 + (size_t)16 * D_;
  for (int k = 0; k < D_; k += 32) {
    bf16x16 A0 = load_a_bf16(a0 + k, D_, lane);
    bf16x16 A1 = load_a_bf16(a1 + k, D_, lane);
#pragma unroll
    for (int t = 0; t < 4; ++t) {
      bf16x16 Bt = load_b_bf16(wbf + (size_t)(n0 + 16 * t) * D_ + k, D_, lane);
      acc[0][t] = wmma_bf16(A0, Bt, acc[0][t]);
      acc[1][t] = wmma_bf16(A1, Bt, acc[1][t]);
    }
  }
  const int hi = lane >> 4, col = lane & 15;
#pragma unroll
  for (int t = 0; t < 4; ++t) {
    int e = n0 + 16 * t + col;
    int which = e >> 10;            // 0=q 1=k 2=v
    int rem = e & 1023;
    int h = rem >> 6, hd = rem & 63;
#pragma unroll
    for (int mi = 0; mi < 2; ++mi) {
#pragma unroll
      for (int r = 0; r < 8; ++r) {
        int m = m0 + 16 * mi + r + 8 * hi;
        int b = m >> 11, s = m & 2047;
        float v = acc[mi][t][r];
        size_t hb = (size_t)(b * H_ + h);
        if (which == 0)
          Q[(hb * S_ + s) * HD_ + hd] = f2bf(v * QSCALE);  // base-2 softmax
        else if (which == 1)
          K[(hb * S_ + s) * HD_ + hd] = f2bf(v);
        else
          Vt[(hb * HD_ + hd) * S_ + s] = f2bf(v);          // transposed V
      }
    }
  }
}

// =============================================================================
// Kernel 2: causal flash attention. One wave owns a 16-query tile of one head.
// 64 keys/iter: 8 WMMAs scores, one online-softmax reduction pass (half-wave
// shuffles), P re-layout via wave-private LDS (+ s_wait_dscnt), 8 WMMAs P*V.
// =============================================================================
__global__ __launch_bounds__(256) void attn_kernel(
    const __bf16* __restrict__ Q, const __bf16* __restrict__ K,
    const __bf16* __restrict__ Vt, __bf16* __restrict__ Ctx) {
  __shared__ __bf16 lds_p[8][16 * 64];    // per-wave 16x64 P staging (2 KiB)
  const int lane = threadIdx.x & 31;
  const int wid = threadIdx.x >> 5;
  const int W = blockIdx.x * 8 + wid;
  const int q0 = (W & 127) * 16;          // S_/16 = 128
  const int h = (W >> 7) & 15;
  const int b = W >> 11;
  const int hi = lane >> 4, col = lane & 15;

  const __bf16* Qb = Q + ((size_t)(b * H_ + h) * S_ + q0) * HD_;
  const __bf16* Kb = K + (size_t)(b * H_ + h) * S_ * HD_;
  const __bf16* Vb = Vt + (size_t)(b * H_ + h) * HD_ * S_;

  bf16x16 qa0 = load_a_bf16(Qb, HD_, lane);        // d = 0..31
  bf16x16 qa1 = load_a_bf16(Qb + 32, HD_, lane);   // d = 32..63

  f32x8 acc[4];
#pragma unroll
  for (int t = 0; t < 4; ++t) acc[t] = zero8();
  float mrow[8], lrow[8];
#pragma unroll
  for (int r = 0; r < 8; ++r) { mrow[r] = -3.0e38f; lrow[r] = 0.0f; }

  __bf16* lp = &lds_p[wid][0];
  const float NEG = -3.0e38f;

  for (int k0 = 0; k0 < q0 + 16; k0 += 64) {
    // ---- scores: 16 queries x 64 keys (already in log2 domain) ----
    f32x8 s[4];
#pragma unroll
    for (int j = 0; j < 4; ++j) {
      s[j] = zero8();
      s[j] = wmma_bf16(qa0, load_b_bf16(Kb + (size_t)(k0 + 16 * j) * HD_, HD_, lane), s[j]);
      s[j] = wmma_bf16(qa1, load_b_bf16(Kb + (size_t)(k0 + 16 * j) * HD_ + 32, HD_, lane), s[j]);
    }

    // ---- causal mask (C layout: row = r + 8*hi, col = lane&15) ----
#pragma unroll
    for (int r = 0; r < 8; ++r) {
      int q = q0 + r + 8 * hi;
#pragma unroll
      for (int j = 0; j < 4; ++j)
        if (k0 + 16 * j + col > q) s[j][r] = NEG;
    }

    // ---- online softmax: one reduction pass per 64 keys ----
#pragma unroll
    for (int r = 0; r < 8; ++r) {
      float t = fmaxf(fmaxf(s[0][r], s[1][r]), fmaxf(s[2][r], s[3][r]));
#pragma unroll
      for (int ms = 1; ms < 16; ms <<= 1) t = fmaxf(t, __shfl_xor(t, ms, 32));
      float mnew = fmaxf(mrow[r], t);
      float alpha = __builtin_amdgcn_exp2f(mrow[r] - mnew);
      mrow[r] = mnew;
      float rs = 0.0f;
#pragma unroll
      for (int j = 0; j < 4; ++j) {
        float p = __builtin_amdgcn_exp2f(s[j][r] - mnew);
        s[j][r] = p;
        rs += p;
      }
#pragma unroll
      for (int ms = 1; ms < 16; ms <<= 1) rs += __shfl_xor(rs, ms, 32);
      lrow[r] = lrow[r] * alpha + rs;
#pragma unroll
      for (int t4 = 0; t4 < 4; ++t4) acc[t4][r] *= alpha;
    }

    // ---- P: C layout -> LDS (row-major 16x64) -> two A operands ----
    asm volatile("" ::: "memory");
#pragma unroll
    for (int r = 0; r < 8; ++r) {
      int row = r + 8 * hi;
#pragma unroll
      for (int j = 0; j < 4; ++j)
        lp[row * 64 + 16 * j + col] = f2bf(s[j][r]);
    }
    asm volatile("s_wait_dscnt 0" ::: "memory");   // stores visible to wave
    bf16x16 pa0 = load_a_bf16(lp, 64, lane);       // key cols 0..31
    bf16x16 pa1 = load_a_bf16(lp + 32, 64, lane);  // key cols 32..63
    asm volatile("s_wait_dscnt 0" ::: "memory");   // loads done before overwrite

    // ---- ctx += P (16x64) * V (64x64): Vt rows are contiguous key runs ----
#pragma unroll
    for (int t4 = 0; t4 < 4; ++t4) {
      acc[t4] = wmma_bf16(pa0, load_b_bf16(Vb + (size_t)(16 * t4) * S_ + k0, S_, lane), acc[t4]);
      acc[t4] = wmma_bf16(pa1, load_b_bf16(Vb + (size_t)(16 * t4) * S_ + k0 + 32, S_, lane), acc[t4]);
    }
  }

  // ---- normalize (rcp) and store ctx[b, s, h*64 + hd] as bf16 ----
  float inv[8];
#pragma unroll
  for (int r = 0; r < 8; ++r) inv[r] = __builtin_amdgcn_rcpf(lrow[r]);
#pragma unroll
  for (int t4 = 0; t4 < 4; ++t4) {
#pragma unroll
    for (int r = 0; r < 8; ++r) {
      int q = q0 + r + 8 * hi;
      Ctx[(size_t)(b * S_ + q) * D_ + h * HD_ + 16 * t4 + col] = f2bf(acc[t4][r] * inv[r]);
    }
  }
}

// =============================================================================
// Kernel 3: output projection, 32x64 tile per wave. out = ctx*w_proj^T + bias
// =============================================================================
__global__ __launch_bounds__(256) void proj_gemm_kernel(
    const __bf16* __restrict__ Ctx, const __bf16* __restrict__ wbf,
    const float* __restrict__ bproj, float* __restrict__ out) {
  const int lane = threadIdx.x & 31;
  const int W = blockIdx.x * 8 + (threadIdx.x >> 5);
  const int m0 = (W & 127) * 32;
  const int n0 = (W >> 7) * 64;    // 16 n-groups
  f32x8 acc[2][4];
#pragma unroll
  for (int i = 0; i < 2; ++i)
#pragma unroll
    for (int t = 0; t < 4; ++t) acc[i][t] = zero8();

  const __bf16* a0 = Ctx + (size_t)m0 * D_;
  const __bf16* a1 = a0 + (size_t)16 * D_;
  for (int k = 0; k < D_; k += 32) {
    bf16x16 A0 = load_a_bf16(a0 + k, D_, lane);
    bf16x16 A1 = load_a_bf16(a1 + k, D_, lane);
#pragma unroll
    for (int t = 0; t < 4; ++t) {
      bf16x16 Bt = load_b_bf16(wbf + (size_t)(n0 + 16 * t) * D_ + k, D_, lane);
      acc[0][t] = wmma_bf16(A0, Bt, acc[0][t]);
      acc[1][t] = wmma_bf16(A1, Bt, acc[1][t]);
    }
  }
  const int hi = lane >> 4, col = lane & 15;
#pragma unroll
  for (int t = 0; t < 4; ++t) {
    int e = n0 + 16 * t + col;
    float bias = bproj[e];
#pragma unroll
    for (int mi = 0; mi < 2; ++mi) {
#pragma unroll
      for (int r = 0; r < 8; ++r) {
        int m = m0 + 16 * mi + r + 8 * hi;
        out[(size_t)m * D_ + e] = acc[mi][t][r] + bias;
      }
    }
  }
}

extern "C" void kernel_launch(void* const* d_in, const int* in_sizes, int n_in,
                              void* d_out, int out_size, void* d_ws, size_t ws_size,
                              hipStream_t stream) {
  (void)in_sizes; (void)n_in; (void)out_size; (void)ws_size;
  const float* x     = (const float*)d_in[0];
  const float* wqkv  = (const float*)d_in[1];
  const float* wproj = (const float*)d_in[2];
  const float* bproj = (const float*)d_in[3];
  float* out = (float*)d_out;

  const size_t E   = (size_t)B_ * H_ * S_ * HD_;  // 4,194,304
  const size_t NX  = (size_t)M_ * D_;             // 4,194,304
  const size_t NW1 = (size_t)3 * D_ * D_;         // 3,145,728
  const size_t NW2 = (size_t)D_ * D_;             // 1,048,576

  __bf16* Q    = (__bf16*)d_ws;    // 8 MiB
  __bf16* K    = Q + E;            // 8 MiB
  __bf16* Vt   = K + E;            // 8 MiB
  __bf16* Ctx  = Vt + E;           // 8 MiB
  __bf16* Xbf  = Ctx + E;          // 8 MiB
  __bf16* Wqkv = Xbf + NX;         // 6 MiB
  __bf16* Wprj = Wqkv + NW1;       // 2 MiB   (48 MiB total)

  cvt_kernel<<<dim3((int)(NX / 8 / 256)), 256, 0, stream>>>(x, Xbf, (int)(NX / 8));
  cvt_kernel<<<dim3((int)(NW1 / 8 / 256)), 256, 0, stream>>>(wqkv, Wqkv, (int)(NW1 / 8));
  cvt_kernel<<<dim3((int)(NW2 / 8 / 256)), 256, 0, stream>>>(wproj, Wprj, (int)(NW2 / 8));

  // 128 m-tiles * 48 n-groups = 6144 waves / 8 per block
  qkv_gemm_kernel<<<dim3(768), 256, 0, stream>>>(Xbf, Wqkv, Q, K, Vt);
  // B*H*S/16 = 4096 waves
  attn_kernel<<<dim3(512), 256, 0, stream>>>(Q, K, Vt, Ctx);
  // 128 * 16 = 2048 waves
  proj_gemm_kernel<<<dim3(256), 256, 0, stream>>>(Ctx, Wprj, bproj, out);
}
